// MambaCritic_80058190398117
// MI455X (gfx1250) — compile-verified, matrix operand order checked by reading, not run on hardware
//
#include <hip/hip_runtime.h>
#include <hip/hip_bf16.h>
#include <cstdint>
#include <cstddef>

#define D_MODEL 512
#define D_INNER 1024
#define D_STATE 16
#define D_CONV  4
#define DT_RANK 32
#define SEQ     512
#define BATCH   4
#define S_DIM   96
#define A_DIM   32
#define LSEQ    511                  // mamba sequence length (SEQ-1)
#define MROWS   (BATCH * LSEQ)       // 2044 rows in all mamba GEMMs

typedef __attribute__((ext_vector_type(16))) __bf16 v16bf;
typedef __attribute__((ext_vector_type(8)))  __bf16 v8bf;
typedef __attribute__((ext_vector_type(8)))  float  v8f;

__device__ __forceinline__ float sigmoidf_(float x) { return 1.f / (1.f + __expf(-x)); }
__device__ __forceinline__ float siluf_(float x)    { return x * sigmoidf_(x); }
__device__ __forceinline__ float softplusf_(float x){ return x > 20.f ? x : log1pf(__expf(x)); }

// Per ISA 7.12.2, a lane's 16 bf16 fragment elements (A or B operand) are two
// contiguous 8-element runs: [k0+half*8, +8) and [k0+half*8+16, +8).
// Both runs are 16-byte aligned when the row base is 16B aligned and lda%8==0.
__device__ __forceinline__ v16bf load_frag16(const __bf16* __restrict__ row, int kk)
{
    const v8bf lo = *(const v8bf*)(row + kk);
    const v8bf hi = *(const v8bf*)(row + kk + 16);
    return __builtin_shufflevector(lo, hi, 0,1,2,3,4,5,6,7,8,9,10,11,12,13,14,15);
}

// ---------------------------------------------------------------------------
// WMMA GEMM:  C[M,N] = act( A[M,K(lda)] * W[N,K]^T + bias[N] ), optional bf16
// side-output Cb. A, W are pre-staged bf16; f32 accumulation.
// Each wave computes a 32x64 strip (8 accumulators; A and B fragments reused).
// ACT: 0 = identity, 1 = softplus.
// Requires K % 32 == 0, N % 64 == 0; M ragged (guarded).
// ---------------------------------------------------------------------------
template <int ACT>
__global__ __launch_bounds__(256)
void wmma_gemm_kernel(const __bf16* __restrict__ A, int lda,
                      const __bf16* __restrict__ W,
                      const float* __restrict__ bias,
                      float* __restrict__ C,
                      __bf16* __restrict__ Cb,
                      int M, int N, int K)
{
    const int wave = threadIdx.x >> 5;
    const int lane = threadIdx.x & 31;
    const int half = lane >> 4;     // 0: lanes 0-15, 1: lanes 16-31
    const int i16  = lane & 15;

    const int nTM = (M + 31) >> 5;       // 32-row strips
    const int nTN = N >> 6;              // 64-col strips
    const int tile = blockIdx.x * 8 + wave;
    if (tile >= nTM * nTN) return;       // wave-uniform: EXEC stays all-1s
    const int tm = tile / nTN;
    const int tn = tile % nTN;

    const int am0 = tm * 32 + i16;
    const int am1 = am0 + 16;
    const __bf16* arow0 = A + (size_t)(am0 < M ? am0 : M - 1) * lda;  // clamp: rows >= M never stored
    const __bf16* arow1 = A + (size_t)(am1 < M ? am1 : M - 1) * lda;
    const __bf16* wrow[4];
    #pragma unroll
    for (int q = 0; q < 4; ++q)
        wrow[q] = W + (size_t)(tn * 64 + q * 16 + i16) * K;

    v8f acc[8] = {};

    for (int k0 = 0; k0 < K; k0 += 32) {
        const int kk = k0 + half * 8;
        const v16bf a0 = load_frag16(arow0, kk);
        const v16bf a1 = load_frag16(arow1, kk);
        #pragma unroll
        for (int q = 0; q < 4; ++q) {
            const v16bf bq = load_frag16(wrow[q], kk);
            acc[q]     = __builtin_amdgcn_wmma_f32_16x16x32_bf16(false, a0, false, bq, (short)0, acc[q],     false, false);
            acc[4 + q] = __builtin_amdgcn_wmma_f32_16x16x32_bf16(false, a1, false, bq, (short)0, acc[4 + q], false, false);
        }
    }

    // C/D layout: element r of lane -> m_local = r + 8*half, n_local = lane&15
    #pragma unroll
    for (int ai = 0; ai < 2; ++ai) {
        #pragma unroll
        for (int q = 0; q < 4; ++q) {
            const int n  = tn * 64 + q * 16 + i16;
            const float bv = bias ? bias[n] : 0.f;
            #pragma unroll
            for (int r = 0; r < 8; ++r) {
                const int m = tm * 32 + ai * 16 + r + 8 * half;
                if (m < M) {
                    float v = acc[ai * 4 + q][r] + bv;
                    if (ACT == 1) v = softplusf_(v);
                    C[(size_t)m * N + n] = v;
                    if (Cb) Cb[(size_t)m * N + n] = (__bf16)v;
                }
            }
        }
    }
}

// ---------------------------------------------------------------------------
// f32 -> bf16 conversion (one-time weight staging)
// ---------------------------------------------------------------------------
__global__ void cvt_bf16_kernel(const float* __restrict__ in, __bf16* __restrict__ out, int n)
{
    const int i = blockIdx.x * blockDim.x + threadIdx.x;
    if (i < n) out[i] = (__bf16)in[i];
}

// ---------------------------------------------------------------------------
// Pack concat(state, action) (SEQ,B,F) -> Xcat[b][l][128] as bf16
// ---------------------------------------------------------------------------
__global__ void pack_input_kernel(const float* __restrict__ state,
                                  const float* __restrict__ action,
                                  __bf16* __restrict__ Xcat)
{
    const int idx = blockIdx.x * blockDim.x + threadIdx.x;
    const int total = BATCH * SEQ * (S_DIM + A_DIM);
    if (idx >= total) return;
    const int f = idx & 127;
    const int l = (idx >> 7) & (SEQ - 1);
    const int b = idx >> 16;   // 128*512 = 2^16
    float v;
    if (f < S_DIM) v = state[(size_t)(l * BATCH + b) * S_DIM + f];
    else           v = action[(size_t)(l * BATCH + b) * A_DIM + (f - S_DIM)];
    Xcat[idx] = (__bf16)v;
}

// ---------------------------------------------------------------------------
// Trim + convert: Hin[b*511+l][h] = bf16(H0[b*512+l][h])
// ---------------------------------------------------------------------------
__global__ void trim_kernel(const float* __restrict__ H0, __bf16* __restrict__ Hin)
{
    const int idx = blockIdx.x * blockDim.x + threadIdx.x;
    if (idx >= MROWS * D_MODEL) return;
    const int h   = idx & (D_MODEL - 1);
    const int row = idx >> 9;
    const int b   = row / LSEQ;
    const int l   = row % LSEQ;
    Hin[idx] = (__bf16)H0[(size_t)(b * SEQ + l) * D_MODEL + h];
}

// ---------------------------------------------------------------------------
// Depthwise causal conv1d (k=4, left pad 3) on u = xz[:, :1024], +bias, SiLU.
// Dual output: f32 (scan) and bf16 (x_proj GEMM operand).
// ---------------------------------------------------------------------------
__global__ void conv_silu_kernel(const float* __restrict__ xz,
                                 const float* __restrict__ conv_w,
                                 const float* __restrict__ conv_b,
                                 float* __restrict__ u2f,
                                 __bf16* __restrict__ u2h)
{
    const int idx = blockIdx.x * blockDim.x + threadIdx.x;
    if (idx >= MROWS * D_INNER) return;
    const int d   = idx & (D_INNER - 1);
    const int row = idx >> 10;
    const int b   = row / LSEQ;
    const int l   = row % LSEQ;
    float s = conv_b[d];
    #pragma unroll
    for (int j = 0; j < D_CONV; ++j) {
        const int ls = l - (D_CONV - 1) + j;
        if (ls >= 0)
            s += conv_w[d * D_CONV + j] * xz[(size_t)(b * LSEQ + ls) * (2 * D_INNER) + d];
    }
    const float v = siluf_(s);
    u2f[idx] = v;
    u2h[idx] = (__bf16)v;
}

// ---------------------------------------------------------------------------
// Selective scan: one thread per (b, d); h[16] in registers, 511 serial steps.
// Fuses +u*D and *silu(z); emits bf16 directly (only consumer is out_proj GEMM).
// xdbl row layout: [0:32)=dt, [32:48)=B, [48:64)=C
// ---------------------------------------------------------------------------
__global__ void scan_kernel(const float* __restrict__ delta,
                            const float* __restrict__ u2f,
                            const float* __restrict__ xdbl,
                            const float* __restrict__ xz,
                            const float* __restrict__ A_log,
                            const float* __restrict__ Dp,
                            __bf16* __restrict__ ybh)
{
    const int t = blockIdx.x * blockDim.x + threadIdx.x;
    if (t >= BATCH * D_INNER) return;
    const int b = t >> 10;
    const int d = t & (D_INNER - 1);

    float Areg[D_STATE], h[D_STATE];
    #pragma unroll
    for (int n = 0; n < D_STATE; ++n) {
        Areg[n] = -__expf(A_log[d * D_STATE + n]);
        h[n] = 0.f;
    }
    const float dskip = Dp[d];

    for (int l = 0; l < LSEQ; ++l) {
        const size_t row = (size_t)(b * LSEQ + l);
        const float dl = delta[row * D_INNER + d];
        const float uu = u2f[row * D_INNER + d];
        const float du = dl * uu;
        const float* xd = xdbl + row * 64;
        float y = 0.f;
        #pragma unroll
        for (int n = 0; n < D_STATE; ++n) {
            h[n] = __expf(dl * Areg[n]) * h[n] + du * xd[32 + n];
            y += h[n] * xd[48 + n];
        }
        const float zz = xz[row * (2 * D_INNER) + D_INNER + d];
        ybh[row * D_INNER + d] = (__bf16)((y + uu * dskip) * siluf_(zz));
    }
}

// ---------------------------------------------------------------------------
// Row softmax over 512 elements, in-place safe; optional bf16 side output
// (feeds the next mamba's in_proj GEMM).
// ---------------------------------------------------------------------------
__global__ __launch_bounds__(256)
void softmax512_kernel(const float* __restrict__ in, float* __restrict__ out,
                       __bf16* __restrict__ outh)
{
    __shared__ float red[256];
    const int row = blockIdx.x;
    const int t = threadIdx.x;
    const float* p = in + (size_t)row * D_MODEL;
    float v0 = p[t], v1 = p[t + 256];
    red[t] = fmaxf(v0, v1);
    __syncthreads();
    for (int s = 128; s > 0; s >>= 1) { if (t < s) red[t] = fmaxf(red[t], red[t + s]); __syncthreads(); }
    const float mx = red[0];
    __syncthreads();
    const float e0 = __expf(v0 - mx), e1 = __expf(v1 - mx);
    red[t] = e0 + e1;
    __syncthreads();
    for (int s = 128; s > 0; s >>= 1) { if (t < s) red[t] += red[t + s]; __syncthreads(); }
    const float inv = 1.f / red[0];
    const float r0 = e0 * inv, r1 = e1 * inv;
    out[(size_t)row * D_MODEL + t]       = r0;
    out[(size_t)row * D_MODEL + t + 256] = r1;
    if (outh) {
        outh[(size_t)row * D_MODEL + t]       = (__bf16)r0;
        outh[(size_t)row * D_MODEL + t + 256] = (__bf16)r1;
    }
}

// ---------------------------------------------------------------------------
// diff + fc3 + leaky_relu + transpose:  out[l][b], shape (511, 4)
//   l < 510 : diff = M1o[b][l+1] - x1[b][l]
//   l == 510: diff = x2[b][509]  - x1[b][510]
// ---------------------------------------------------------------------------
__global__ __launch_bounds__(256)
void final_kernel(const float* __restrict__ M1o,
                  const float* __restrict__ x1,
                  const float* __restrict__ x2,
                  const float* __restrict__ fc3_w,
                  const float* __restrict__ fc3_b,
                  float* __restrict__ out)
{
    __shared__ float red[256];
    const int row = blockIdx.x;          // b*511 + l
    const int b = row / LSEQ;
    const int l = row % LSEQ;
    const int t = threadIdx.x;
    const size_t r1 = (size_t)row * D_MODEL;
    const size_t rp = (l < LSEQ - 1) ? (size_t)(b * LSEQ + l + 1) * D_MODEL
                                     : (size_t)(b * LSEQ + (LSEQ - 2)) * D_MODEL;
    const float* pos = (l < LSEQ - 1) ? M1o : x2;
    float partial = 0.f;
    for (int h = t; h < D_MODEL; h += 256)
        partial += (pos[rp + h] - x1[r1 + h]) * fc3_w[h];
    red[t] = partial;
    __syncthreads();
    for (int s = 128; s > 0; s >>= 1) { if (t < s) red[t] += red[t + s]; __syncthreads(); }
    if (t == 0) {
        float v = red[0] + fc3_b[0];
        out[l * BATCH + b] = v > 0.f ? v : 0.01f * v;   // leaky_relu
    }
}

// ---------------------------------------------------------------------------
// Host-side orchestration
// ---------------------------------------------------------------------------
struct MambaW {
    // bf16-staged GEMM weights
    const __bf16 *in_proj_wh, *x_proj_wh, *dt_proj_wh, *out_proj_wh;
    // f32 aux params
    const float *conv_w, *conv_b, *dt_proj_b, *A_log, *Dp;
};

static inline void launch_gemm(const __bf16* A, int lda, const __bf16* W, const float* bias,
                               float* C, __bf16* Cb, int M, int N, int K, int act, hipStream_t s)
{
    const int tiles  = ((M + 31) / 32) * (N / 64);
    const int blocks = (tiles + 7) / 8;
    if (act == 1)
        wmma_gemm_kernel<1><<<blocks, 256, 0, s>>>(A, lda, W, bias, C, Cb, M, N, K);
    else
        wmma_gemm_kernel<0><<<blocks, 256, 0, s>>>(A, lda, W, bias, C, Cb, M, N, K);
}

static inline void cvt_bf16(const float* in, __bf16* out, int n, hipStream_t s)
{
    cvt_bf16_kernel<<<(n + 255) / 256, 256, 0, s>>>(in, out, n);
}

static inline void run_mamba(const MambaW& p, const __bf16* Xin, float* Out, __bf16* Outh,
                             float* xz, float* u2f, __bf16* u2h, float* xdbl, __bf16* xdblh,
                             float* delta, __bf16* ybh, hipStream_t s)
{
    // in_proj: [2044,512] x [2048,512]^T -> xz (f32)
    launch_gemm(Xin, D_MODEL, p.in_proj_wh, nullptr, xz, nullptr, MROWS, 2 * D_INNER, D_MODEL, 0, s);
    {
        const int n = MROWS * D_INNER;
        conv_silu_kernel<<<(n + 255) / 256, 256, 0, s>>>(xz, p.conv_w, p.conv_b, u2f, u2h);
    }
    // x_proj: [2044,1024] x [64,1024]^T -> xdbl (f32 for scan, bf16 for dt_proj)
    launch_gemm(u2h, D_INNER, p.x_proj_wh, nullptr, xdbl, xdblh, MROWS, 64, D_INNER, 0, s);
    // dt_proj (+bias, softplus): A = xdblh[:, :32] with lda=64
    launch_gemm(xdblh, 64, p.dt_proj_wh, p.dt_proj_b, delta, nullptr, MROWS, D_INNER, DT_RANK, 1, s);
    // selective scan (serial over L), fused D-skip + silu(z) gate, bf16 out
    scan_kernel<<<(BATCH * D_INNER + 255) / 256, 256, 0, s>>>(delta, u2f, xdbl, xz,
                                                              p.A_log, p.Dp, ybh);
    // out_proj: [2044,1024] x [512,1024]^T -> Out (f32) + Outh (bf16, next mamba input)
    launch_gemm(ybh, D_INNER, p.out_proj_wh, nullptr, Out, Outh, MROWS, D_MODEL, D_INNER, 0, s);
}

extern "C" void kernel_launch(void* const* d_in, const int* in_sizes, int n_in,
                              void* d_out, int out_size, void* d_ws, size_t ws_size,
                              hipStream_t stream)
{
    (void)in_sizes; (void)n_in; (void)out_size; (void)ws_size;

    const float* state  = (const float*)d_in[0];
    const float* action = (const float*)d_in[1];
    const float* fc1_w  = (const float*)d_in[2];
    const float* fc1_b  = (const float*)d_in[3];
    const float* fc3_w  = (const float*)d_in[4];
    const float* fc3_b  = (const float*)d_in[5];
    const float* m1w[9]; const float* m2w[9];
    for (int i = 0; i < 9; ++i) { m1w[i] = (const float*)d_in[6 + i]; m2w[i] = (const float*)d_in[15 + i]; }
    // order: in_proj_w, conv_w, conv_b, x_proj_w, dt_proj_w, dt_proj_b, A_log, D, out_proj_w
    float* out = (float*)d_out;

    // -------- workspace carve-out: f32 region, then bf16 region (16B-aligned) ----
    float* fp = (float*)d_ws;
    float* H0    = fp; fp += (size_t)BATCH * SEQ * D_MODEL;     // 1,048,576
    float* xz    = fp; fp += (size_t)MROWS * 2 * D_INNER;       // 4,186,112
    float* u2f   = fp; fp += (size_t)MROWS * D_INNER;           // 2,093,056
    float* xdbl  = fp; fp += (size_t)MROWS * 64;                //   130,816
    float* delta = fp; fp += (size_t)MROWS * D_INNER;           // 2,093,056
    float* M1o   = fp; fp += (size_t)MROWS * D_MODEL;
    float* x1    = fp; fp += (size_t)MROWS * D_MODEL;
    float* x2    = fp; fp += (size_t)MROWS * D_MODEL;

    __bf16* bp = (__bf16*)fp;
    __bf16* Xcath = bp; bp += (size_t)BATCH * SEQ * (S_DIM + A_DIM);
    __bf16* Hinh  = bp; bp += (size_t)MROWS * D_MODEL;
    __bf16* u2h   = bp; bp += (size_t)MROWS * D_INNER;
    __bf16* xdblh = bp; bp += (size_t)MROWS * 64;
    __bf16* ybh   = bp; bp += (size_t)MROWS * D_INNER;
    __bf16* M1oh  = bp; bp += (size_t)MROWS * D_MODEL;
    __bf16* x1h   = bp; bp += (size_t)MROWS * D_MODEL;
    __bf16* fc1_wh = bp; bp += (size_t)D_MODEL * (S_DIM + A_DIM);
    __bf16* wh[2][4];   // per layer: in_proj, x_proj, dt_proj, out_proj
    const int wsz[4] = { 2 * D_INNER * D_MODEL, (DT_RANK + 2 * D_STATE) * D_INNER,
                         D_INNER * DT_RANK, D_MODEL * D_INNER };
    for (int ly = 0; ly < 2; ++ly)
        for (int wi = 0; wi < 4; ++wi) { wh[ly][wi] = bp; bp += (size_t)wsz[wi]; }

    // -------- stage weights to bf16 (once per launch) --------
    cvt_bf16(fc1_w, fc1_wh, D_MODEL * (S_DIM + A_DIM), stream);
    const float* wsrc1[4] = { m1w[0], m1w[3], m1w[4], m1w[8] };
    const float* wsrc2[4] = { m2w[0], m2w[3], m2w[4], m2w[8] };
    for (int wi = 0; wi < 4; ++wi) {
        cvt_bf16(wsrc1[wi], wh[0][wi], wsz[wi], stream);
        cvt_bf16(wsrc2[wi], wh[1][wi], wsz[wi], stream);
    }
    MambaW m1 = { wh[0][0], wh[0][1], wh[0][2], wh[0][3], m1w[1], m1w[2], m1w[5], m1w[6], m1w[7] };
    MambaW m2 = { wh[1][0], wh[1][1], wh[1][2], wh[1][3], m2w[1], m2w[2], m2w[5], m2w[6], m2w[7] };

    // 1) concat(state, action) -> bf16
    {
        const int n = BATCH * SEQ * (S_DIM + A_DIM);
        pack_input_kernel<<<(n + 255) / 256, 256, 0, stream>>>(state, action, Xcath);
    }
    // 2) fc1: H0 = Xcat @ fc1_w^T + fc1_b
    launch_gemm(Xcath, S_DIM + A_DIM, fc1_wh, fc1_b, H0, nullptr,
                BATCH * SEQ, D_MODEL, S_DIM + A_DIM, 0, stream);
    // 3) drop last timestep, convert to bf16
    {
        const int n = MROWS * D_MODEL;
        trim_kernel<<<(n + 255) / 256, 256, 0, stream>>>(H0, Hinh);
    }
    // 4) out = mamba(m1, Hin)                (f32 M1o + bf16 M1oh)
    run_mamba(m1, Hinh, M1o, M1oh, xz, u2f, u2h, xdbl, xdblh, delta, ybh, stream);
    // 5) x1 = softmax(mamba(m1, out))        (bf16 x1h feeds m2)
    run_mamba(m1, M1oh, x1, nullptr, xz, u2f, u2h, xdbl, xdblh, delta, ybh, stream);
    softmax512_kernel<<<MROWS, 256, 0, stream>>>(x1, x1, x1h);
    // 6) x2 = softmax(mamba(m2, x1))
    run_mamba(m2, x1h, x2, nullptr, xz, u2f, u2h, xdbl, xdblh, delta, ybh, stream);
    softmax512_kernel<<<MROWS, 256, 0, stream>>>(x2, x2, nullptr);
    // 7) diff -> fc3 -> leaky_relu -> (511, 4)
    final_kernel<<<MROWS, 256, 0, stream>>>(M1o, x1, x2, fc3_w, fc3_b, out);
}
